// CustomGATLayer_56873956933922
// MI455X (gfx1250) — matrix operand order repaired and checked by reading, not compiled, not run on hardware
//
#include <hip/hip_runtime.h>
#include <hip/hip_bf16.h>
#include <math.h>

typedef __attribute__((ext_vector_type(16))) _Float16 v16h;
typedef __attribute__((ext_vector_type(8)))  float    v8f;

#define NEG_SLOPE 0.01f
// key(-inf): bits(-inf)=0xFF800000, MSB set -> ~0xFF800000 = 0x007FFFFF
#define KEY_NEG_INF 0x007FFFFFu

// Order-preserving map float -> uint so atomicMax works for all signs.
__device__ __forceinline__ unsigned fkey(float f) {
  unsigned u = __float_as_uint(f);
  return (u & 0x80000000u) ? ~u : (u | 0x80000000u);
}
__device__ __forceinline__ float fdec(unsigned k) {
  unsigned u = (k & 0x80000000u) ? (k ^ 0x80000000u) : ~k;
  return __uint_as_float(u);
}

// ---------------------------------------------------------------------------
// K0: out[n][d] = bias[d] (bias folded into accumulator init); s=0; m=-inf
// ---------------------------------------------------------------------------
__global__ __launch_bounds__(256) void gat_init(float* __restrict__ out,
                                                const float* __restrict__ bias,
                                                float* __restrict__ s,
                                                unsigned* __restrict__ mkey,
                                                int N, int D) {
  long long idx = (long long)blockIdx.x * blockDim.x + threadIdx.x;
  long long ND  = (long long)N * D;
  if (idx < ND) out[idx] = bias[(int)(idx % D)];
  if (idx < N) { s[idx] = 0.0f; mkey[idx] = KEY_NEG_INF; }
}

// ---------------------------------------------------------------------------
// K1: projections el = h_src . attn_l, er = h_dst . attn_r via WMMA.
// One wave handles 16 nodes. A-tile layout (16-bit A 16x32, ISA 7.12.2):
//   lanes 0-15 : row M=lane, K = {kb..kb+7, kb+16..kb+23}
//   lanes 16-31: row M=lane-16, K = {kb+8..kb+15, kb+24..kb+31}
// B broadcasts attn over all 16 N-columns:
//   B uses the same K order as A (lanes 0-15: kb..kb+7, kb+16..kb+23; lanes 16-31: kb+8..kb+15, kb+24..kb+31).
// D[m][n] is replicated over n; read from lane 0 (M=0..7) / lane 16 (M=8..15).
// ---------------------------------------------------------------------------
__device__ __forceinline__ void load_a_tile(const float* __restrict__ row,
                                            int hi, int kb, v16h& a) {
  int base0 = kb + (hi ? 8 : 0);        // K for VGPRs 0-3 (elements 0..7)
  int base1 = kb + 16 + (hi ? 8 : 0);   // K for VGPRs 4-7 (elements 8..15)
  const float4* p0 = (const float4*)(row + base0);
  const float4* p1 = (const float4*)(row + base1);
  float4 x0 = p0[0], x1 = p0[1], y0 = p1[0], y1 = p1[1];
  a[0]  = (_Float16)x0.x; a[1]  = (_Float16)x0.y; a[2]  = (_Float16)x0.z; a[3]  = (_Float16)x0.w;
  a[4]  = (_Float16)x1.x; a[5]  = (_Float16)x1.y; a[6]  = (_Float16)x1.z; a[7]  = (_Float16)x1.w;
  a[8]  = (_Float16)y0.x; a[9]  = (_Float16)y0.y; a[10] = (_Float16)y0.z; a[11] = (_Float16)y0.w;
  a[12] = (_Float16)y1.x; a[13] = (_Float16)y1.y; a[14] = (_Float16)y1.z; a[15] = (_Float16)y1.w;
}

__global__ __launch_bounds__(256) void gat_proj_wmma(const float* __restrict__ hsrc,
                                                     const float* __restrict__ hdst,
                                                     const float* __restrict__ attn_l,
                                                     const float* __restrict__ attn_r,
                                                     float* __restrict__ el,
                                                     float* __restrict__ er,
                                                     int N, int D) {
  int lane  = threadIdx.x & 31;
  int gwave = blockIdx.x * (blockDim.x >> 5) + (threadIdx.x >> 5);
  int node0 = gwave * 16;
  if (node0 >= N) return;
  int hi  = lane >> 4;
  int m   = lane & 15;
  int row = node0 + m; if (row >= N) row = N - 1;
  const float* rs = hsrc + (long long)row * D;
  const float* rd = hdst + (long long)row * D;

  v8f cl = {}; v8f cr = {};
  for (int kb = 0; kb < D; kb += 32) {
    int kb2 = kb + (hi ? 8 : 0);
    v16h a, b;
    #pragma unroll
    for (int i = 0; i < 16; ++i) b[i] = (_Float16)attn_l[kb2 + i + (i & 8)];
    load_a_tile(rs, hi, kb, a);
    cl = __builtin_amdgcn_wmma_f32_16x16x32_f16(false, a, false, b,
                                                (short)0, cl, false, false);
    #pragma unroll
    for (int i = 0; i < 16; ++i) b[i] = (_Float16)attn_r[kb2 + i + (i & 8)];
    load_a_tile(rd, hi, kb, a);
    cr = __builtin_amdgcn_wmma_f32_16x16x32_f16(false, a, false, b,
                                                (short)0, cr, false, false);
  }
  if (lane == 0 || lane == 16) {       // N-column 0 of each half holds M=0..7 / 8..15
    int mb = node0 + (hi ? 8 : 0);
    #pragma unroll
    for (int r = 0; r < 8; ++r) {
      int n = mb + r;
      if (n < N) { el[n] = cl[r]; er[n] = cr[r]; }
    }
  }
}

// ---------------------------------------------------------------------------
// K2: e = leaky_relu(el[src] + er[dst]); segment-max via encoded atomicMax
// ---------------------------------------------------------------------------
__global__ __launch_bounds__(256) void gat_score(const int* __restrict__ esrc,
                                                 const int* __restrict__ edst,
                                                 const float* __restrict__ el,
                                                 const float* __restrict__ er,
                                                 float* __restrict__ e,
                                                 unsigned* __restrict__ mkey,
                                                 int E) {
  int i = blockIdx.x * blockDim.x + threadIdx.x;
  if (i >= E) return;
  int d = edst[i];
  float v = el[esrc[i]] + er[d];
  v = (v > 0.0f) ? v : NEG_SLOPE * v;
  e[i] = v;
  atomicMax(&mkey[d], fkey(v));
}

// ---------------------------------------------------------------------------
// K3: a = exp(e - m[dst]) (stored back over e); segment-sum via atomicAdd
// ---------------------------------------------------------------------------
__global__ __launch_bounds__(256) void gat_expsum(const int* __restrict__ edst,
                                                  float* __restrict__ e,
                                                  const unsigned* __restrict__ mkey,
                                                  float* __restrict__ s,
                                                  int E) {
  int i = blockIdx.x * blockDim.x + threadIdx.x;
  if (i >= E) return;
  int d = edst[i];
  float a = __expf(e[i] - fdec(mkey[d]));
  e[i] = a;
  atomicAdd(&s[d], a);
}

// ---------------------------------------------------------------------------
// K4: out[dst] += (a/s[dst]) * h_src[src]; one wave per edge, float4 per lane.
// h_src and out are L2-resident (51 MB each, 192 MB L2) -> L2 atomic bound.
// ---------------------------------------------------------------------------
__global__ __launch_bounds__(256) void gat_scatter(const float* __restrict__ hsrc,
                                                   const int* __restrict__ esrc,
                                                   const int* __restrict__ edst,
                                                   const float* __restrict__ a,
                                                   const float* __restrict__ s,
                                                   float* __restrict__ out,
                                                   int E, int D) {
  int lane = threadIdx.x & 31;
  int ge   = blockIdx.x * (blockDim.x >> 5) + (threadIdx.x >> 5);
  if (ge >= E) return;
  int src = esrc[ge], dst = edst[ge];
  float alpha = a[ge] / s[dst];
  int d0 = lane * 4;
  if (d0 + 3 < D) {
    float4 v = *(const float4*)(hsrc + (long long)src * D + d0);
    float* op = out + (long long)dst * D + d0;
    atomicAdd(op + 0, alpha * v.x);
    atomicAdd(op + 1, alpha * v.y);
    atomicAdd(op + 2, alpha * v.z);
    atomicAdd(op + 3, alpha * v.w);
  }
}

// ---------------------------------------------------------------------------
extern "C" void kernel_launch(void* const* d_in, const int* in_sizes, int n_in,
                              void* d_out, int out_size, void* d_ws, size_t ws_size,
                              hipStream_t stream) {
  const float* hsrc   = (const float*)d_in[0];
  const float* hdst   = (const float*)d_in[1];
  const int*   esrc   = (const int*)d_in[2];
  const int*   edst   = (const int*)d_in[3];
  const float* attn_l = (const float*)d_in[4];
  const float* attn_r = (const float*)d_in[5];
  const float* bias   = (const float*)d_in[6];
  int D = in_sizes[4];           // 128
  int N = in_sizes[0] / D;       // 100000
  int E = in_sizes[2];           // 1000000
  float* out = (float*)d_out;

  // workspace layout (≈5.6 MB): el[N] er[N] mkey[N] s[N] e[E]
  float*    el   = (float*)d_ws;
  float*    er   = el + N;
  unsigned* mkey = (unsigned*)(er + N);
  float*    s    = (float*)(mkey + N);
  float*    e    = s + N;

  long long ND = (long long)N * D;
  gat_init<<<(int)((ND + 255) / 256), 256, 0, stream>>>(out, bias, s, mkey, N, D);

  int tiles = (N + 15) / 16;                 // 16 nodes per wave
  gat_proj_wmma<<<(tiles + 7) / 8, 256, 0, stream>>>(hsrc, hdst, attn_l, attn_r,
                                                     el, er, N, D);

  gat_score <<<(E + 255) / 256, 256, 0, stream>>>(esrc, edst, el, er, e, mkey, E);
  gat_expsum<<<(E + 255) / 256, 256, 0, stream>>>(edst, e, mkey, s, E);
  gat_scatter<<<(E + 7) / 8, 256, 0, stream>>>(hsrc, esrc, edst, e, s, out, E, D);
}
